// NodeConvGNN_21096879358622
// MI455X (gfx1250) — matrix operand order repaired and checked by reading, not compiled, not run on hardware
//
#include <hip/hip_runtime.h>

typedef __attribute__((ext_vector_type(16))) __bf16 v16bf;
typedef __attribute__((ext_vector_type(8)))  float  v8f;

#define DFEAT 128

// round-to-nearest-even f32 -> bf16 (only used in bandwidth-bound elementwise/pack kernels)
static __device__ __forceinline__ __bf16 f2bf(float f) {
    union { float f; unsigned u; } x; x.f = f;
    unsigned r = (x.u + 0x7FFFu + ((x.u >> 16) & 1u)) >> 16;
    unsigned short h = (unsigned short)r;
    __bf16 o;
    __builtin_memcpy(&o, &h, 2);
    return o;
}

// 16 contiguous bf16 (32B, aligned) -> v16bf  (two b128 loads; global or LDS)
static __device__ __forceinline__ v16bf load_btile(const __bf16* p) {
    struct alignas(32) { uint4 a, b; } t;
    const uint4* q = reinterpret_cast<const uint4*>(p);
    t.a = q[0];
    t.b = q[1];
    v16bf r;
    __builtin_memcpy(&r, &t, 32);
    return r;
}

// A fragment for one K-step: 8 bf16 at p, 8 bf16 at p+16 (per ISA 16-bit A layout)
static __device__ __forceinline__ v16bf load_atile(const __bf16* p) {
    struct alignas(32) { uint4 a, b; } t;
    t.a = *reinterpret_cast<const uint4*>(p);
    t.b = *reinterpret_cast<const uint4*>(p + 16);
    v16bf r;
    __builtin_memcpy(&r, &t, 32);
    return r;
}

// ---------------- degree / norm ----------------
__global__ void deg_init_kernel(float* deg, int n) {
    int i = blockIdx.x * blockDim.x + threadIdx.x;
    if (i < n) deg[i] = 1.0f;                      // self-loop
}
__global__ void deg_count_kernel(float* deg, const int* __restrict__ dst, int e) {
    int i = blockIdx.x * blockDim.x + threadIdx.x;
    if (i < e) atomicAdd(deg + dst[i], 1.0f);
}
__global__ void deg_fin_kernel(float* dinv, int n) {
    int i = blockIdx.x * blockDim.x + threadIdx.x;
    if (i < n) dinv[i] = rsqrtf(dinv[i]);          // deg >= 1
}

// ---------------- f32 -> bf16 copy (for x) ----------------
__global__ void cvt_bf16_kernel(const float* __restrict__ in, __bf16* __restrict__ out,
                                long long total) {
    long long i = (long long)blockIdx.x * blockDim.x + threadIdx.x;
    if (i < total) out[i] = f2bf(in[i]);
}

// ---------------- weight pre-swizzle into WMMA B layout (bf16) ----------------
// B element j of lane l for tile (nt, ks):  n = nt*16 + (l&15),  k = ks*32 + (l>>4)*8 + (j&7) + (j>>3)*16
__global__ void pack_w_kernel(const float* __restrict__ W, __bf16* __restrict__ Wp, int ksteps) {
    int tid = blockIdx.x * blockDim.x + threadIdx.x;
    int total = 8 * ksteps * 32 * 16;
    if (tid >= total) return;
    int j    = tid & 15;
    int lane = (tid >> 4) & 31;
    int t    = tid >> 9;             // nt*ksteps + ks
    int ks   = t % ksteps;
    int nt   = t / ksteps;
    int n    = nt * 16 + (lane & 15);
    int k    = ks * 32 + (lane >> 4) * 8 + (j & 7) + (j >> 3) * 16;
    Wp[tid] = f2bf(W[(size_t)k * DFEAT + n]);
}

// ---------------- node-level GEMM: Y[N,128](f32) = Xb[N,128](bf16) @ W ----------------
__global__ __launch_bounds__(256) void gemm_node_kernel(
    const __bf16* __restrict__ X, const __bf16* __restrict__ Wp,
    float* __restrict__ Y, int nrows)
{
    const int wave = threadIdx.x >> 5;
    const int lane = threadIdx.x & 31;
    const int half = lane >> 4;
    const int rowBase = blockIdx.x * 128 + wave * 16;
    int m  = rowBase + (lane & 15);
    int mc = m < nrows ? m : nrows - 1;            // clamp loads, keep EXEC all-ones
    const __bf16* row = X + (size_t)mc * DFEAT;

    v16bf a[4];
#pragma unroll
    for (int ks = 0; ks < 4; ++ks)
        a[ks] = load_atile(row + ks * 32 + half * 8);

    const v8f zero = {0.f, 0.f, 0.f, 0.f, 0.f, 0.f, 0.f, 0.f};
    const bool fullTile = (rowBase + 16) <= nrows;  // wave-uniform
#pragma unroll
    for (int nt = 0; nt < 8; ++nt) {
        v8f acc = zero;
#pragma unroll
        for (int ks = 0; ks < 4; ++ks) {
            v16bf b = load_btile(Wp + ((size_t)((nt * 4 + ks) * 32 + lane)) * 16);
            acc = __builtin_amdgcn_wmma_f32_16x16x32_bf16(
                false, a[ks], false, b, (short)0, acc, false, false);
        }
        const int n = nt * 16 + (lane & 15);
        if (fullTile) {
            float* col = Y + (size_t)(rowBase + half * 8) * DFEAT + n;
#pragma unroll
            for (int r = 0; r < 8; ++r) col[(size_t)r * DFEAT] = acc[r];
        } else {
#pragma unroll
            for (int r = 0; r < 8; ++r) {           // C/D layout: M = r + half*8
                int mr = rowBase + half * 8 + r;
                if (mr < nrows) Y[(size_t)mr * DFEAT + n] = acc[r];
            }
        }
    }
}

// ---------------- self-loop init: A[i] = H[i] * dinv[i]^2 ----------------
__global__ void self_init_kernel(const float* __restrict__ H, const float* __restrict__ dinv,
                                 float* __restrict__ A, int n) {
    int tid = blockIdx.x * blockDim.x + threadIdx.x;
    if (tid >= n * 32) return;
    int i = tid >> 5, c = tid & 31;
    float di = dinv[i];
    float s = di * di;
    float4 v = *reinterpret_cast<const float4*>(H + (size_t)i * DFEAT + c * 4);
    float4 o; o.x = v.x * s; o.y = v.y * s; o.z = v.z * s; o.w = v.w * s;
    *reinterpret_cast<float4*>(A + (size_t)i * DFEAT + c * 4) = o;
}

// ---------------- edge scatter: A[dst] += H[src] * dinv[src]*dinv[dst] ----------------
__global__ void edge_scatter_kernel(const float* __restrict__ H, float* __restrict__ A,
                                    const int* __restrict__ src, const int* __restrict__ dst,
                                    const float* __restrict__ dinv, int e) {
    long long tid = (long long)blockIdx.x * blockDim.x + threadIdx.x;
    if (tid >= (long long)e * 32) return;
    int ei = (int)(tid >> 5), c = (int)(tid & 31);
    int s = src[ei], d = dst[ei];
    float nm = dinv[s] * dinv[d];
    float4 v = *reinterpret_cast<const float4*>(H + (size_t)s * DFEAT + c * 4);
    float* base = A + (size_t)d * DFEAT + c * 4;
    atomicAdd(base + 0, v.x * nm);
    atomicAdd(base + 1, v.y * nm);
    atomicAdd(base + 2, v.z * nm);
    atomicAdd(base + 3, v.w * nm);
}

// ---------------- bias + relu, emit bf16 node features ----------------
__global__ void bias_relu_bf16_kernel(const float* __restrict__ A, const float* __restrict__ b,
                                      __bf16* __restrict__ H, long long total) {
    long long tid = (long long)blockIdx.x * blockDim.x + threadIdx.x;
    if (tid >= total) return;
    int c = (int)(tid & (DFEAT - 1));
    float v = A[tid] + b[c];
    H[tid] = f2bf(v > 0.f ? v : 0.f);
}

// ---------------- fused edge MLP: relu([h[src]|h[dst]] @ W1 + b1) @ W2 + b2 ----------------
// W1 packed (64KB) is staged to LDS once per block; A operands are bf16 row gathers.
__global__ __launch_bounds__(256) void edge_mlp_kernel(
    const __bf16* __restrict__ H, const int* __restrict__ src, const int* __restrict__ dst,
    const __bf16* __restrict__ W1p, const float* __restrict__ b1,
    const float* __restrict__ W2, const float* __restrict__ b2,
    float* __restrict__ out, int E)
{
    extern __shared__ uint4 s4[];                  // 4096 * 16B = 64KB dynamic LDS
    const uint4* g4 = reinterpret_cast<const uint4*>(W1p);
    for (int i = threadIdx.x; i < 4096; i += 256) s4[i] = g4[i];
    __syncthreads();
    const __bf16* sW = reinterpret_cast<const __bf16*>(s4);

    const int wave = threadIdx.x >> 5;
    const int lane = threadIdx.x & 31;
    const int half = lane >> 4;
    const int tileBase = blockIdx.x * 128 + wave * 16;
    int e  = tileBase + (lane & 15);
    int ec = e < E ? e : E - 1;                    // clamp, keep EXEC all-ones
    const int s = src[ec], d = dst[ec];
    const __bf16* rs = H + (size_t)s * DFEAT;
    const __bf16* rd = H + (size_t)d * DFEAT;

    v16bf a[8];                                    // K = 256 -> 8 K-steps of 32
#pragma unroll
    for (int ks = 0; ks < 8; ++ks) {
        const __bf16* row = (ks < 4) ? rs : rd;
        a[ks] = load_atile(row + (ks & 3) * 32 + half * 8);
    }

    const int n = lane & 15;
    float bl[8], w2[8];
#pragma unroll
    for (int nt = 0; nt < 8; ++nt) { bl[nt] = b1[nt * 16 + n]; w2[nt] = W2[nt * 16 + n]; }

    const v8f zero = {0.f, 0.f, 0.f, 0.f, 0.f, 0.f, 0.f, 0.f};
    v8f accs[8];
#pragma unroll
    for (int nt = 0; nt < 8; ++nt) {
        v8f acc = zero;
#pragma unroll
        for (int ks = 0; ks < 8; ++ks) {
            v16bf b = load_btile(sW + ((size_t)((nt * 8 + ks) * 32 + lane)) * 16);
            acc = __builtin_amdgcn_wmma_f32_16x16x32_bf16(
                false, a[ks], false, b, (short)0, acc, false, false);
        }
        accs[nt] = acc;
    }

    // lin2: per-row dot over n; row m = r + half*8 lives in lanes (half*16 .. half*16+15)
#pragma unroll
    for (int r = 0; r < 8; ++r) {
        float sum = 0.f;
#pragma unroll
        for (int nt = 0; nt < 8; ++nt) {
            float v = accs[nt][r] + bl[nt];
            v = v > 0.f ? v : 0.f;
            sum += v * w2[nt];
        }
        sum += __shfl_xor(sum, 1, 32);
        sum += __shfl_xor(sum, 2, 32);
        sum += __shfl_xor(sum, 4, 32);
        sum += __shfl_xor(sum, 8, 32);
        int eo = tileBase + half * 8 + r;
        if (n == 0 && eo < E) out[eo] = sum + b2[0];
    }
}

// ---------------- launch ----------------
extern "C" void kernel_launch(void* const* d_in, const int* in_sizes, int n_in,
                              void* d_out, int out_size, void* d_ws, size_t ws_size,
                              hipStream_t stream) {
    (void)n_in; (void)out_size; (void)ws_size;
    const float* x   = (const float*)d_in[0];
    const int*   gei = (const int*)d_in[1];
    // d_in[2] (lg_edge_index) is unused by the reference
    const float* Wc1 = (const float*)d_in[3];
    const float* bc1 = (const float*)d_in[4];
    const float* Wc2 = (const float*)d_in[5];
    const float* bc2 = (const float*)d_in[6];
    const float* Wl1 = (const float*)d_in[7];
    const float* bl1 = (const float*)d_in[8];
    const float* Wl2 = (const float*)d_in[9];
    const float* bl2 = (const float*)d_in[10];
    float* out = (float*)d_out;

    const int N = in_sizes[0] / DFEAT;
    const int E = in_sizes[1] / 2;
    const int* src = gei;
    const int* dst = gei + E;
    const long long NF = (long long)N * DFEAT;

    auto align_up = [](size_t v) { return (v + 255) & ~(size_t)255; };
    char* w = (char*)d_ws;
    float*  dinv = (float*)w;  w += align_up((size_t)N * 4);
    float*  bufA = (float*)w;  w += align_up((size_t)N * DFEAT * 4);   // gemm out (f32)
    float*  bufB = (float*)w;  w += align_up((size_t)N * DFEAT * 4);   // aggregation (f32)
    __bf16* hbuf = (__bf16*)w; w += align_up((size_t)N * DFEAT * 2);   // bf16 features (x, h1, h2)
    __bf16* Wc1p = (__bf16*)w; w += align_up((size_t)8 * 4 * 32 * 16 * 2);
    __bf16* Wc2p = (__bf16*)w; w += align_up((size_t)8 * 4 * 32 * 16 * 2);
    __bf16* Wl1p = (__bf16*)w; w += align_up((size_t)8 * 8 * 32 * 16 * 2);

    const dim3 blk(256);
    const int nBlkN    = (N + 255) / 256;
    const int nBlkE    = (E + 255) / 256;
    const int gemmBlks = (N + 127) / 128;
    const int mlpBlks  = (E + 127) / 128;
    const int selfBlks = (int)(((long long)N * 32 + 255) / 256);
    const int scatBlks = (int)(((long long)E * 32 + 255) / 256);
    const int nfBlks   = (int)((NF + 255) / 256);

    // degrees + normalization
    deg_init_kernel <<<nBlkN, blk, 0, stream>>>(dinv, N);
    deg_count_kernel<<<nBlkE, blk, 0, stream>>>(dinv, dst, E);
    deg_fin_kernel  <<<nBlkN, blk, 0, stream>>>(dinv, N);

    // pre-swizzle weights into WMMA B layout (bf16)
    pack_w_kernel<<<(8 * 4 * 32 * 16 + 255) / 256, blk, 0, stream>>>(Wc1, Wc1p, 4);
    pack_w_kernel<<<(8 * 4 * 32 * 16 + 255) / 256, blk, 0, stream>>>(Wc2, Wc2p, 4);
    pack_w_kernel<<<(8 * 8 * 32 * 16 + 255) / 256, blk, 0, stream>>>(Wl1, Wl1p, 8);

    // x -> bf16
    cvt_bf16_kernel<<<nfBlks, blk, 0, stream>>>(x, hbuf, NF);

    // layer 1: h1 = relu(Agg(x @ W1) + b1)
    gemm_node_kernel    <<<gemmBlks, blk, 0, stream>>>(hbuf, Wc1p, bufA, N);
    self_init_kernel    <<<selfBlks, blk, 0, stream>>>(bufA, dinv, bufB, N);
    edge_scatter_kernel <<<scatBlks, blk, 0, stream>>>(bufA, bufB, src, dst, dinv, E);
    bias_relu_bf16_kernel<<<nfBlks,  blk, 0, stream>>>(bufB, bc1, hbuf, NF);

    // layer 2: h2 = relu(Agg(h1 @ W2) + b2)
    gemm_node_kernel    <<<gemmBlks, blk, 0, stream>>>(hbuf, Wc2p, bufA, N);
    self_init_kernel    <<<selfBlks, blk, 0, stream>>>(bufA, dinv, bufB, N);
    edge_scatter_kernel <<<scatBlks, blk, 0, stream>>>(bufA, bufB, src, dst, dinv, E);
    bias_relu_bf16_kernel<<<nfBlks,  blk, 0, stream>>>(bufB, bc2, hbuf, NF);

    // fused edge MLP (dominant GEMM; W1 staged in 64KB LDS, 64 WMMAs per 16-edge wave tile)
    edge_mlp_kernel<<<mlpBlks, blk, 65536, stream>>>(hbuf, src, dst, Wl1p, bl1, Wl2, bl2, out, E);
}
